// Linear2Bit_27273042329840
// MI455X (gfx1250) — matrix operand order repaired and checked by reading, not compiled, not run on hardware
//
#include <hip/hip_runtime.h>

#define IN_F 4096
#define OUT_F 11008
#define GROUPS_PER_ROW (IN_F / 32)   // 128 groups along K per output row
#define KSTEPS (IN_F / 32)           // 128 WMMA K-steps total
#define SPLIT_K 4                    // K split factor (32 steps per wave)
#define KSTEPS_PER_SPLIT (KSTEPS / SPLIT_K)
#define WAVES_PER_BLOCK 4
#define BLOCK_THREADS (WAVES_PER_BLOCK * 32)
#define M_ROWS 16
#define OUT_ELEMS (M_ROWS * OUT_F)   // 176128

typedef _Float16 v16h __attribute__((ext_vector_type(16)));
typedef _Float16 v8h  __attribute__((ext_vector_type(8)));
typedef float    v8f  __attribute__((ext_vector_type(8)));
typedef float    v4f  __attribute__((ext_vector_type(4)));
typedef int      v4i  __attribute__((ext_vector_type(4)));

// Prologue: convert activations to f16 once (reused by every N-tile; L2-resident).
__global__ __launch_bounds__(256) void cvt_x_f16(const float* __restrict__ x,
                                                 _Float16* __restrict__ xh, int n) {
  for (int i = blockIdx.x * blockDim.x + threadIdx.x; i < n;
       i += gridDim.x * blockDim.x)
    xh[i] = (_Float16)x[i];
}

// Main: each wave computes a 16(M) x 16(N) f32 partial tile over 32 K-steps
// (split-K x4 for memory-level parallelism) via chained v_wmma_f32_16x16x32_f16,
// dequantizing 2-bit weights through a 256-entry LDS LUT.
__global__ __launch_bounds__(BLOCK_THREADS) void linear2bit_wmma(
    const _Float16* __restrict__ xh,     // [16, 4096] f16 (from d_ws)
    const int* __restrict__ wq,          // [OUT_F*128, 8] packed 2-bit (low byte of each dword)
    const _Float16* __restrict__ wnorm,  // [OUT_F*128] group scales
    float* __restrict__ partial)         // [SPLIT_K, 16, OUT_F] f32 partials (d_ws)
{
  // 256-entry LUT: byte -> 4 dequant levels (before group scale). 2 KB LDS.
  __shared__ _Float16 s_lut[256][4];
  const int tid = threadIdx.x;
  for (int t = tid; t < 256; t += BLOCK_THREADS) {
    const float lv[4] = {-1.0f, -0.333f, 0.333f, 1.0f};
#pragma unroll
    for (int i = 0; i < 4; ++i)
      s_lut[t][i] = (_Float16)lv[(t >> (2 * i)) & 3];
  }
  __syncthreads();

  const int wave    = tid >> 5;
  const int lane    = tid & 31;
  const int half_id = lane >> 4;   // which 16-lane half of the wave
  const int colidx  = lane & 15;
  const int n = (blockIdx.x * WAVES_PER_BLOCK + wave) * 16 + colidx;  // B column
  const int kg0 = blockIdx.y * KSTEPS_PER_SPLIT;                      // K split start

  // Weight words for this lane: half 0 takes dwords 0..3 (K 0..15 of group),
  // half 1 takes dwords 4..7 (K 16..31)  ->  B element e has K = e + 16*half.
  const v4i* wptr =
      (const v4i*)(wq + (size_t)n * GROUPS_PER_ROW * 8 + half_id * 4) + (size_t)kg0 * 2;
  const _Float16* nrm = wnorm + (size_t)n * GROUPS_PER_ROW + kg0;
  // A row m = colidx; lanes0-15 hold K=0..15, lanes16-31 hold K=16..31
  // (matches B)  ->  two contiguous 8-half (16B) loads.
  const _Float16* ap = xh + (size_t)colidx * IN_F + (size_t)kg0 * 32 + half_id * 16;

  v8f acc = {};

#pragma unroll 2
  for (int kg = 0; kg < KSTEPS_PER_SPLIT; ++kg) {
    // Stream weights ahead of use (speculative, dropped if OOB).
    __builtin_prefetch(&wptr[(kg + 8) * 2], 0, 0);

    // ---- A fragment (16x32 f16) ----
    v8h a0 = *(const v8h*)(ap + kg * 32);        // K = 16*half + 0..7
    v8h a1 = *(const v8h*)(ap + kg * 32 + 8);    // K = 16*half + 8..15
    v16h A;
#pragma unroll
    for (int i = 0; i < 8; ++i) { A[i] = a0[i]; A[8 + i] = a1[i]; }

    // ---- B fragment (32x16 f16) : dequantize 16 codes via LUT ----
    v4i w4 = __builtin_nontemporal_load(&wptr[kg * 2]);  // 4 dwords = 16 codes
    _Float16 hn = nrm[kg];

    v16h B;
#pragma unroll
    for (int wI = 0; wI < 4; ++wI) {
      const int byte = w4[wI] & 0xFF;
#pragma unroll
      for (int i = 0; i < 4; ++i)
        B[4 * wI + i] = s_lut[byte][i] * hn;   // K = 4*wI + i (+16*half)
    }

    acc = __builtin_amdgcn_wmma_f32_16x16x32_f16(
        /*neg_a=*/false, A, /*neg_b=*/false, B,
        /*c_mod=*/(short)0, acc, /*reuse_a=*/false, /*reuse_b=*/false);
  }

  // ---- write f32 partial tile: C/D layout row = r + 8*half, col = colidx ----
  float* pout = partial + (size_t)blockIdx.y * OUT_ELEMS;
#pragma unroll
  for (int r = 0; r < 8; ++r)
    pout[(size_t)(r + 8 * half_id) * OUT_F + n] = acc[r];
}

// Deterministic reduction: out = sum_s partial[s] + bias (fixed order, no atomics).
__global__ __launch_bounds__(256) void reduce_partials(
    const float* __restrict__ partial,  // [SPLIT_K, 16, OUT_F]
    const float* __restrict__ bias,     // [OUT_F]
    float* __restrict__ out)            // [16, OUT_F]
{
  const int i4 = blockIdx.x * blockDim.x + threadIdx.x;  // float4 index
  const int total4 = OUT_ELEMS / 4;                      // 44032
  if (i4 >= total4) return;

  const v4f* p = (const v4f*)partial;
  v4f s = p[i4];
#pragma unroll
  for (int sp = 1; sp < SPLIT_K; ++sp)
    s += p[(size_t)sp * (OUT_ELEMS / 4) + i4];

  const int n4 = i4 % (OUT_F / 4);  // column group for bias
  v4f b = ((const v4f*)bias)[n4];
  ((v4f*)out)[i4] = s + b;
}

extern "C" void kernel_launch(void* const* d_in, const int* in_sizes, int n_in,
                              void* d_out, int out_size, void* d_ws, size_t ws_size,
                              hipStream_t stream) {
  const float*    x    = (const float*)d_in[0];
  const int*      wq   = (const int*)d_in[1];
  const _Float16* wn   = (const _Float16*)d_in[2];
  const float*    bias = (const float*)d_in[3];
  float* out = (float*)d_out;

  // d_ws layout: [0,128KB) f16 activations; then SPLIT_K f32 partial outputs.
  _Float16* xh      = (_Float16*)d_ws;
  float*    partial = (float*)((char*)d_ws + (size_t)M_ROWS * IN_F * sizeof(_Float16));

  cvt_x_f16<<<64, 256, 0, stream>>>(x, xh, M_ROWS * IN_F);

  // N: 11008 / (4 waves * 16 cols) = 172 blocks; K: 4 splits -> 688 blocks, 2752 waves.
  dim3 grid(OUT_F / (WAVES_PER_BLOCK * 16), SPLIT_K);
  linear2bit_wmma<<<grid, BLOCK_THREADS, 0, stream>>>(xh, wq, wn, partial);

  reduce_partials<<<(OUT_ELEMS / 4 + 255) / 256, 256, 0, stream>>>(partial, bias, out);
}
